// GCNModelAE_social_61933428417199
// MI455X (gfx1250) — compile-verified, matrix operand order checked by reading, not run on hardware
//
#include <hip/hip_runtime.h>
#include <hip/hip_bf16.h>

// ---------------------------------------------------------------------------
// CDNA5 (gfx1250) implementation of the GCN+GAT autoencoder forward pass.
//   * all heavy matmuls via v_wmma_f32_16x16x32_bf16 (bf16 operands, f32 acc)
//   * GAT attention fused flash-style (no [8,N,N] coefs materialization)
//   * 32x64 C-tile per wave (two A fragments share every B fragment -> half
//     the L2 B-traffic), explicit ping-pong software pipeline in K.
// ---------------------------------------------------------------------------

typedef __bf16 bf16;
typedef __attribute__((ext_vector_type(16))) __bf16 v16bf;
typedef __attribute__((ext_vector_type(8)))  __bf16 v8bf;
typedef __attribute__((ext_vector_type(8)))  float  v8f;

// Load a 16x32 bf16 WMMA A/B fragment; q = row/lane-resolved base + k offset.
// lane 0-15 : row = lane,    K = +0..7 (v0-3) and +16..23 (v4-7)
// lane16-31 : row = lane-16, K = +8..15 (v0-3) and +24..31 (v4-7)
__device__ __forceinline__ v16bf frag_at(const bf16* __restrict__ q) {
  v8bf lo = *(const v8bf*)(q);
  v8bf hi = *(const v8bf*)(q + 16);
  v16bf f;
#pragma unroll
  for (int e = 0; e < 8; ++e) { f[e] = lo[e]; f[e + 8] = hi[e]; }
  return f;
}

// ---------------------------------------------------------------------------
// f32 -> bf16 conversion, optional transpose / relu / zero-padding, batched.
// dst: bf16 [batch][Rp][Cp].  transpose==0: src [batch][R][C]
//                             transpose==1: src [batch][C][R]  (dst = src^T)
// ---------------------------------------------------------------------------
__global__ void k_conv_bf16(const float* __restrict__ src, long sstride,
                            bf16* __restrict__ dst, long dstride,
                            int R, int Rp, int C, int Cp, int transpose, int relu) {
  long total = (long)Rp * Cp;
  long i = blockIdx.x * (long)blockDim.x + threadIdx.x;
  int b = blockIdx.y;
  if (i >= total) return;
  int r = (int)(i / Cp), c = (int)(i % Cp);
  float v = 0.f;
  if (r < R && c < C)
    v = transpose ? src[(long)b * sstride + (long)c * R + r]
                  : src[(long)b * sstride + (long)r * C + c];
  if (relu) v = fmaxf(v, 0.f);
  dst[(long)b * dstride + i] = (bf16)v;
}

// ---------------------------------------------------------------------------
// Generic WMMA GEMM:  C[b] = act( A[b] @ BT[b]^T + bias )
//   A : bf16 [M][lda]        lda = Kp, K multiple of 64, M multiple of 32
//   BT: bf16 [rup64(N)][ldb] (row n = column n of B, zero rows beyond N)
//   C : f32  [M][ldc]; optional Cbf (bf16 copy), CTbf (bf16 transposed copy)
// One wave per 32x64 C tile; two A fragments reuse each B fragment;
// ping-pong pipelined K loop (next 32-K slab loads issued before WMMAs).
// ---------------------------------------------------------------------------
struct Frags { v16bf a0, a1, b0, b1, b2, b3; };

__device__ __forceinline__ Frags load_set(const bf16* ap0, const bf16* ap1,
                                          const bf16* bp0, const bf16* bp1,
                                          const bf16* bp2, const bf16* bp3,
                                          long k) {
  Frags f;
  f.a0 = frag_at(ap0 + k);
  f.a1 = frag_at(ap1 + k);
  f.b0 = frag_at(bp0 + k);
  f.b1 = frag_at(bp1 + k);
  f.b2 = frag_at(bp2 + k);
  f.b3 = frag_at(bp3 + k);
  return f;
}

__device__ __forceinline__ void mma_set(const Frags& f, v8f acc[2][4]) {
  acc[0][0] = __builtin_amdgcn_wmma_f32_16x16x32_bf16(false, f.a0, false, f.b0, (short)0, acc[0][0], false, false);
  acc[0][1] = __builtin_amdgcn_wmma_f32_16x16x32_bf16(false, f.a0, false, f.b1, (short)0, acc[0][1], false, false);
  acc[0][2] = __builtin_amdgcn_wmma_f32_16x16x32_bf16(false, f.a0, false, f.b2, (short)0, acc[0][2], false, false);
  acc[0][3] = __builtin_amdgcn_wmma_f32_16x16x32_bf16(false, f.a0, false, f.b3, (short)0, acc[0][3], false, false);
  acc[1][0] = __builtin_amdgcn_wmma_f32_16x16x32_bf16(false, f.a1, false, f.b0, (short)0, acc[1][0], false, false);
  acc[1][1] = __builtin_amdgcn_wmma_f32_16x16x32_bf16(false, f.a1, false, f.b1, (short)0, acc[1][1], false, false);
  acc[1][2] = __builtin_amdgcn_wmma_f32_16x16x32_bf16(false, f.a1, false, f.b2, (short)0, acc[1][2], false, false);
  acc[1][3] = __builtin_amdgcn_wmma_f32_16x16x32_bf16(false, f.a1, false, f.b3, (short)0, acc[1][3], false, false);
}

__global__ void __launch_bounds__(32)
k_gemm_wmma(const bf16* __restrict__ A, long lda, long strideA,
            const bf16* __restrict__ BT, long ldb, long strideB,
            float* __restrict__ C, long ldc, long strideC,
            const float* __restrict__ bias,
            bf16* __restrict__ Cbf, long ldcbf, long strideCbf,
            bf16* __restrict__ CTbf, long ldct, long strideCT,
            int M, int N, int K, int relu) {
  int lane = threadIdx.x & 31;
  long mt = (long)blockIdx.x * 32;
  long nt0 = (long)blockIdx.y * 64;
  int b = blockIdx.z;
  int rA = lane & 15;
  int half8 = (lane >> 4) * 8;

  // per-lane fragment base pointers (row + K-half offsets folded in)
  const bf16* ap0 = A + (long)b * strideA + (mt + rA) * lda + half8;
  const bf16* ap1 = ap0 + 16 * lda;
  const bf16* bp0 = BT + (long)b * strideB + (nt0 + 0  + rA) * ldb + half8;
  const bf16* bp1 = BT + (long)b * strideB + (nt0 + 16 + rA) * ldb + half8;
  const bf16* bp2 = BT + (long)b * strideB + (nt0 + 32 + rA) * ldb + half8;
  const bf16* bp3 = BT + (long)b * strideB + (nt0 + 48 + rA) * ldb + half8;

  v8f acc[2][4] = {};
  // ping-pong pipeline over 32-wide K slabs (K is a multiple of 64)
  Frags c0 = load_set(ap0, ap1, bp0, bp1, bp2, bp3, 0);
  long k = 32;
  for (; k + 32 < K; k += 64) {
    __builtin_prefetch((const void*)(ap0 + k + 1024), 0, 0);
    Frags c1 = load_set(ap0, ap1, bp0, bp1, bp2, bp3, k);
    mma_set(c0, acc);
    c0 = load_set(ap0, ap1, bp0, bp1, bp2, bp3, k + 32);
    mma_set(c1, acc);
  }
  Frags c1 = load_set(ap0, ap1, bp0, bp1, bp2, bp3, k);
  mma_set(c0, acc);
  mma_set(c1, acc);

  int mo = (lane >> 4) * 8;
#pragma unroll
  for (int g = 0; g < 2; ++g) {
#pragma unroll
    for (int t = 0; t < 4; ++t) {
      long col = nt0 + t * 16 + rA;
      if (col >= N) continue;
      float bv = bias ? bias[col] : 0.f;
#pragma unroll
      for (int r = 0; r < 8; ++r) {
        long row = mt + g * 16 + mo + r;
        float v = acc[g][t][r] + bv;
        if (relu) v = fmaxf(v, 0.f);
        C[(long)b * strideC + row * ldc + col] = v;
        if (Cbf)  Cbf[(long)b * strideCbf + row * ldcbf + col] = (bf16)v;
        if (CTbf) CTbf[(long)b * strideCT + col * ldct + row] = (bf16)v;
      }
    }
  }
}

// ---------------------------------------------------------------------------
// f1[h][n] = fts[h][n][:].a1[h]   f2[h][n] = fts[h][n][:].a2[h]
// ---------------------------------------------------------------------------
__global__ void k_f1f2(const float* __restrict__ fts, const float* __restrict__ a1,
                       const float* __restrict__ a2, float* __restrict__ f1,
                       float* __restrict__ f2, int N) {
  int h = blockIdx.y;
  long n = blockIdx.x * (long)blockDim.x + threadIdx.x;
  if (n >= N) return;
  const float* p = fts + ((long)h * N + n) * 64;
  const float* A1 = a1 + h * 64;
  const float* A2 = a2 + h * 64;
  float s1 = 0.f, s2 = 0.f;
#pragma unroll 8
  for (int o = 0; o < 64; ++o) {
    float v = p[o];
    s1 = fmaf(v, A1[o], s1);
    s2 = fmaf(v, A2[o], s2);
  }
  f1[(long)h * N + n] = s1;
  f2[(long)h * N + n] = s2;
}

// ---------------------------------------------------------------------------
// Fused flash-GAT:  out = act( softmax(leaky(f1+f2)+bias) @ fts + b + resid )
// one wave per (head, 32-row tile).  Phase 1: online (max,sum) per row.
// Phase 2: regenerate exp tiles as two bf16 A-fragments sharing the loaded
// B-fragments of ftsT (8 WMMAs per 32-wide j tile).
//   mode 0: relu, write to h1 [N][512] at col h*64 (+bf16 copy)
//   mode 1: identity, write to hB [8][N][64]
// ---------------------------------------------------------------------------
__global__ void __launch_bounds__(32)
k_gat_flash(const float* __restrict__ f1, const float* __restrict__ f2,
            const float* __restrict__ biasmat, const bf16* __restrict__ ftsT,
            const float* __restrict__ resid, const float* __restrict__ headb,
            float* __restrict__ out_f32, long ldo,
            bf16* __restrict__ out_bf, long ldobf, int N, int mode) {
  __shared__ float sm[32], ss[32];
  int lane = threadIdx.x & 31;
  int h = blockIdx.y;
  long rowbase = (long)blockIdx.x * 32;
  const float* F1 = f1 + (long)h * N;
  const float* F2 = f2 + (long)h * N;

  // ---- phase 1: per-row online softmax stats (32 lanes cooperate per row)
  for (int r = 0; r < 32; ++r) {
    long row = rowbase + r;
    float fr = F1[row];
    const float* br = biasmat + row * (long)N;
    float m = -1e30f, s = 0.f;
    for (int j = lane; j < N; j += 32) {
      float l = fr + F2[j];
      l = (l > 0.f) ? l : 0.2f * l;
      l += br[j];
      if (l > m) { s = s * __expf(m - l) + 1.f; m = l; }
      else        s += __expf(l - m);
    }
#pragma unroll
    for (int o = 16; o >= 1; o >>= 1) {
      float m2 = __shfl_xor(m, o);
      float s2 = __shfl_xor(s, o);
      float mn = fmaxf(m, m2);
      s = s * __expf(m - mn) + s2 * __expf(m2 - mn);
      m = mn;
    }
    if (lane == 0) { sm[r] = m; ss[r] = s; }
  }
  __syncthreads();

  // ---- phase 2: P~ @ V with WMMA (two 16-row P tiles share the V fragments)
  int rl = lane & 15;
  int half8 = (lane >> 4) * 8;
  float m0 = sm[rl],       m1 = sm[16 + rl];
  float fr0 = F1[rowbase + rl], fr1 = F1[rowbase + 16 + rl];
  const float* br0 = biasmat + (rowbase + rl) * (long)N;
  const float* br1 = biasmat + (rowbase + 16 + rl) * (long)N;
  const bf16* vt0 = ftsT + (long)h * 64 * N + (long)(0  + rl) * N + half8;
  const bf16* vt1 = ftsT + (long)h * 64 * N + (long)(16 + rl) * N + half8;
  const bf16* vt2 = ftsT + (long)h * 64 * N + (long)(32 + rl) * N + half8;
  const bf16* vt3 = ftsT + (long)h * 64 * N + (long)(48 + rl) * N + half8;

  v8f acc[2][4] = {};
  for (int jt = 0; jt < N; jt += 32) {
    v16bf vf0 = frag_at(vt0 + jt);
    v16bf vf1 = frag_at(vt1 + jt);
    v16bf vf2 = frag_at(vt2 + jt);
    v16bf vf3 = frag_at(vt3 + jt);
    v16bf pf0, pf1;
#pragma unroll
    for (int e = 0; e < 8; ++e) {
      int j0 = jt + half8 + e;
      int j1 = j0 + 16;
      float fa = F2[j0], fb = F2[j1];
      float l;
      l = fr0 + fa; l = (l > 0.f) ? l : 0.2f * l; l += br0[j0];
      pf0[e] = (bf16)__expf(l - m0);
      l = fr0 + fb; l = (l > 0.f) ? l : 0.2f * l; l += br0[j1];
      pf0[e + 8] = (bf16)__expf(l - m0);
      l = fr1 + fa; l = (l > 0.f) ? l : 0.2f * l; l += br1[j0];
      pf1[e] = (bf16)__expf(l - m1);
      l = fr1 + fb; l = (l > 0.f) ? l : 0.2f * l; l += br1[j1];
      pf1[e + 8] = (bf16)__expf(l - m1);
    }
    acc[0][0] = __builtin_amdgcn_wmma_f32_16x16x32_bf16(false, pf0, false, vf0, (short)0, acc[0][0], false, false);
    acc[0][1] = __builtin_amdgcn_wmma_f32_16x16x32_bf16(false, pf0, false, vf1, (short)0, acc[0][1], false, false);
    acc[0][2] = __builtin_amdgcn_wmma_f32_16x16x32_bf16(false, pf0, false, vf2, (short)0, acc[0][2], false, false);
    acc[0][3] = __builtin_amdgcn_wmma_f32_16x16x32_bf16(false, pf0, false, vf3, (short)0, acc[0][3], false, false);
    acc[1][0] = __builtin_amdgcn_wmma_f32_16x16x32_bf16(false, pf1, false, vf0, (short)0, acc[1][0], false, false);
    acc[1][1] = __builtin_amdgcn_wmma_f32_16x16x32_bf16(false, pf1, false, vf1, (short)0, acc[1][1], false, false);
    acc[1][2] = __builtin_amdgcn_wmma_f32_16x16x32_bf16(false, pf1, false, vf2, (short)0, acc[1][2], false, false);
    acc[1][3] = __builtin_amdgcn_wmma_f32_16x16x32_bf16(false, pf1, false, vf3, (short)0, acc[1][3], false, false);
  }

  int mo = (lane >> 4) * 8;
  const float* HB = headb + h * 64;
#pragma unroll
  for (int g = 0; g < 2; ++g) {
#pragma unroll
    for (int t = 0; t < 4; ++t) {
      int o = t * 16 + rl;
      float bv = HB[o];
#pragma unroll
      for (int r = 0; r < 8; ++r) {
        long row = rowbase + g * 16 + mo + r;
        float v = acc[g][t][r] / ss[g * 16 + mo + r] + bv +
                  resid[((long)h * N + row) * 64 + o];
        if (mode == 0) {
          v = fmaxf(v, 0.f);
          out_f32[row * ldo + h * 64 + o] = v;
          out_bf[row * ldobf + h * 64 + o] = (bf16)v;
        } else {
          out_f32[((long)h * N + row) * 64 + o] = v;
        }
      }
    }
  }
}

// mean over 8 heads of hB[8][N][64] -> dst[N][ldd] at column coloff (+bf copy)
__global__ void k_mean_heads(const float* __restrict__ hB, float* __restrict__ dst,
                             long ldd, bf16* __restrict__ dstbf, long lddbf,
                             long coloff, int N) {
  long i = blockIdx.x * (long)blockDim.x + threadIdx.x;
  if (i >= (long)N * 64) return;
  long row = i >> 6;
  int o = (int)(i & 63);
  float s = 0.f;
#pragma unroll
  for (int h = 0; h < 8; ++h) s += hB[((long)h * N + row) * 64 + o];
  s *= 0.125f;
  dst[row * ldd + coloff + o] = s;
  dstbf[row * lddbf + coloff + o] = (bf16)s;
}

// me[i] = concat(latent[GID1[i]], cancat2[GID2[i]]) as bf16 [M][128]
__global__ void k_gather_me(const float* __restrict__ latent,
                            const float* __restrict__ cancat2,
                            const int* __restrict__ gid1, const int* __restrict__ gid2,
                            bf16* __restrict__ mebf, int M) {
  long i = blockIdx.x * (long)blockDim.x + threadIdx.x;
  if (i >= (long)M * 128) return;
  long row = i >> 7;
  int c = (int)(i & 127);
  float v = (c < 64) ? latent[(long)gid1[row] * 64 + c]
                     : cancat2[(long)gid2[row] * 64 + (c - 64)];
  mebf[i] = (bf16)v;
}

// ===========================================================================
// Host orchestration
// ===========================================================================
namespace {

inline int rup(int x, int a) { return (x + a - 1) / a * a; }

struct Ctx {
  hipStream_t stream;
  char* ws;
  size_t off;
  void* take(size_t b) {
    off = (off + 255) & ~(size_t)255;
    void* p = ws + off;
    off += b;
    return p;
  }
};

void conv(Ctx& c, const float* src, long sstride, bf16* dst, long dstride,
          int R, int Rp, int C, int Cp, int tr, int relu, int batch) {
  long total = (long)Rp * Cp;
  dim3 g((unsigned)((total + 255) / 256), batch);
  k_conv_bf16<<<g, dim3(256), 0, c.stream>>>(src, sstride, dst, dstride,
                                             R, Rp, C, Cp, tr, relu);
}

// BT buffers always padded (zero rows) to a multiple of 64 rows;
// K a multiple of 64; M a multiple of 32.
void gemm(Ctx& c, const bf16* A, long lda, long sA, const bf16* BT, long ldb, long sB,
          float* C, long ldc, long sC, const float* bias,
          bf16* Cbf, long ldcbf, long sCbf, bf16* CTbf, long ldct, long sCT,
          int M, int N, int K, int relu, int batch) {
  int Npad = rup(N, 64);
  dim3 g((unsigned)(M / 32), (unsigned)(Npad / 64), batch);
  k_gemm_wmma<<<g, dim3(32), 0, c.stream>>>(
      A, lda, sA, BT, ldb, sB, C, ldc, sC, bias,
      Cbf, ldcbf, sCbf, CTbf, ldct, sCT, M, N, K, relu);
}

struct SpBufs {  // shared scratch for all 6 spgat invocations (max Fin = 2048)
  bf16 *WT, *WresT, *ftsT, *h1bf, *BWT, *BWresT, *fts2T;
  float *fts, *resid, *f1, *f2, *h1, *fts2, *resid2, *f1b, *f2b, *hB;
};

void spgat(Ctx& c, SpBufs& S, const bf16* xbf, int Fin, int Finp,
           const float* biasmat, const float* const* gp,
           float* dst, bf16* dstbf, long ldd, long coloff, int N) {
  // gp: 0 A_W 1 A_Wres 2 A_a1 3 A_a2 4 A_b 5 B_W 6 B_Wres 7 B_a1 8 B_a2 9 B_b
  conv(c, gp[0], (long)Fin * 64, S.WT, 64L * Finp, 64, 64, Fin, Finp, 1, 0, 8);
  conv(c, gp[1], (long)Fin * 64, S.WresT, 64L * Finp, 64, 64, Fin, Finp, 1, 0, 8);
  gemm(c, xbf, Finp, 0, S.WT, Finp, 64L * Finp, S.fts, 64, (long)N * 64,
       nullptr, nullptr, 0, 0, S.ftsT, N, 64L * N, N, 64, Finp, 0, 8);
  gemm(c, xbf, Finp, 0, S.WresT, Finp, 64L * Finp, S.resid, 64, (long)N * 64,
       nullptr, nullptr, 0, 0, nullptr, 0, 0, N, 64, Finp, 0, 8);
  dim3 gf((unsigned)((N + 255) / 256), 8);
  k_f1f2<<<gf, dim3(256), 0, c.stream>>>(S.fts, gp[2], gp[3], S.f1, S.f2, N);
  k_gat_flash<<<dim3((unsigned)(N / 32), 8), dim3(32), 0, c.stream>>>(
      S.f1, S.f2, biasmat, S.ftsT, S.resid, gp[4], S.h1, 512L, S.h1bf, 512L, N, 0);

  conv(c, gp[5], 512L * 64, S.BWT, 64L * 512, 64, 64, 512, 512, 1, 0, 8);
  conv(c, gp[6], 512L * 64, S.BWresT, 64L * 512, 64, 64, 512, 512, 1, 0, 8);
  gemm(c, S.h1bf, 512, 0, S.BWT, 512, 64L * 512, S.fts2, 64, (long)N * 64,
       nullptr, nullptr, 0, 0, S.fts2T, N, 64L * N, N, 64, 512, 0, 8);
  gemm(c, S.h1bf, 512, 0, S.BWresT, 512, 64L * 512, S.resid2, 64, (long)N * 64,
       nullptr, nullptr, 0, 0, nullptr, 0, 0, N, 64, 512, 0, 8);
  k_f1f2<<<gf, dim3(256), 0, c.stream>>>(S.fts2, gp[7], gp[8], S.f1b, S.f2b, N);
  k_gat_flash<<<dim3((unsigned)(N / 32), 8), dim3(32), 0, c.stream>>>(
      S.f1b, S.f2b, biasmat, S.fts2T, S.resid2, gp[9], S.hB, 0L, nullptr, 0L, N, 1);
  k_mean_heads<<<dim3((unsigned)(((long)N * 64 + 255) / 256)), dim3(256), 0, c.stream>>>(
      S.hB, dst, ldd, dstbf, ldd, coloff, N);
}

void gcn(Ctx& c, const bf16* xbf, int Fin, int Finp, const float* W,
         const bf16* adjbf, float* dst, bf16* dstbf, long ldd, int relu, int N,
         bf16* WTs, float* ts, bf16* tTs) {
  conv(c, W, 0, WTs, 0, 64, 64, Fin, Finp, 1, 0, 1);
  gemm(c, xbf, Finp, 0, WTs, Finp, 0, ts, 64, 0, nullptr,
       nullptr, 0, 0, tTs, N, 0, N, 64, Finp, 0, 1);
  gemm(c, adjbf, N, 0, tTs, N, 0, dst, ldd, 0, nullptr,
       dstbf, ldd, 0, nullptr, 0, 0, N, 64, N, relu, 1);
}

struct GParams {
  const float *Wc, *Wgc1, *Wgc2, *Wgc3, *bc;
  const float* gat1[10];
  const float* gat2[10];
  const float* gat3[10];
};

void branch(Ctx& c, SpBufs& S, const float* feat, const float* adj,
            const float* biasmat, const GParams& g, int N, int F,
            float* recon_dst, float** emb_ret, bf16** embbf_ret, float** cancat_ret) {
  int Fp = rup(F, 64);
  bf16* xbf = (bf16*)c.take((size_t)N * Fp * 2);
  bf16* adjbf = (bf16*)c.take((size_t)N * N * 2);
  float* l1 = (float*)c.take((size_t)N * 128 * 4);
  bf16* l1bf = (bf16*)c.take((size_t)N * 128 * 2);
  float* l2 = (float*)c.take((size_t)N * 128 * 4);
  bf16* l2bf = (bf16*)c.take((size_t)N * 128 * 2);
  float* emb = (float*)c.take((size_t)N * 128 * 4);
  bf16* embbf = (bf16*)c.take((size_t)N * 128 * 2);
  bf16* embrelu = (bf16*)c.take((size_t)N * 128 * 2);
  float* cancat = (float*)c.take((size_t)N * 64 * 4);
  bf16* cancatbf = (bf16*)c.take((size_t)N * 64 * 2);
  bf16* WcT = (bf16*)c.take((size_t)64 * 128 * 2);
  bf16* WgT = (bf16*)c.take((size_t)64 * Fp * 2);
  float* t = (float*)c.take((size_t)N * 64 * 4);
  bf16* tT = (bf16*)c.take((size_t)64 * N * 2);

  conv(c, feat, 0, xbf, 0, N, N, F, Fp, 0, 0, 1);
  conv(c, adj, 0, adjbf, 0, N, N, N, N, 0, 0, 1);

  gcn(c, xbf, F, Fp, g.Wgc1, adjbf, l1, l1bf, 128, 1, N, WgT, t, tT);
  spgat(c, S, xbf, F, Fp, biasmat, g.gat1, l1, l1bf, 128, 64, N);
  gcn(c, l1bf, 128, 128, g.Wgc2, adjbf, l2, l2bf, 128, 1, N, WgT, t, tT);
  spgat(c, S, l1bf, 128, 128, biasmat, g.gat2, l2, l2bf, 128, 64, N);
  gcn(c, l2bf, 128, 128, g.Wgc3, adjbf, emb, embbf, 128, 0, N, WgT, t, tT);
  spgat(c, S, l2bf, 128, 128, biasmat, g.gat3, emb, embbf, 128, 64, N);

  conv(c, emb, 0, embrelu, 0, N, N, 128, 128, 0, 1, 1);
  conv(c, g.Wc, 0, WcT, 0, 64, 64, 128, 128, 1, 0, 1);
  gemm(c, embrelu, 128, 0, WcT, 128, 0, cancat, 64, 0, g.bc,
       cancatbf, 64, 0, nullptr, 0, 0, N, 64, 128, 0, 1);
  // recon = cancat @ cancat^T  (A and BT are the same row-major buffer)
  gemm(c, cancatbf, 64, 0, cancatbf, 64, 0, recon_dst, N, 0, nullptr,
       nullptr, 0, 0, nullptr, 0, 0, N, N, 64, 0, 1);

  *emb_ret = emb;
  *embbf_ret = embbf;
  *cancat_ret = cancat;
}

GParams load_g(void* const* d_in, int base) {
  GParams g;
  auto P = [&](int i) { return (const float*)d_in[base + i]; };
  g.Wc = P(0); g.Wgc1 = P(1); g.Wgc2 = P(2); g.Wgc3 = P(3); g.bc = P(4);
  for (int i = 0; i < 10; ++i) g.gat1[i] = P(5 + i);
  for (int i = 0; i < 10; ++i) g.gat2[i] = P(15 + i);
  for (int i = 0; i < 10; ++i) g.gat3[i] = P(25 + i);
  return g;
}

}  // namespace

extern "C" void kernel_launch(void* const* d_in, const int* in_sizes, int n_in,
                              void* d_out, int out_size, void* d_ws, size_t ws_size,
                              hipStream_t stream) {
  (void)in_sizes; (void)n_in; (void)out_size; (void)ws_size;
  const int N1 = 2048, N2 = 2048, F1 = 2000, F2 = 1800, M = 512;

  // Inputs: setup_inputs() insertion order, params pytree flattened with
  // sorted dict keys (jax convention):
  //  0 feat1  1 adj1  2 bias1  3 feat2  4 adj2  5 bias2  6 GID1  7 GID2
  //  8 Wd1  9 Wf  10 Wlm  11 Wo  12 bd1  13 bf  14 blm  15 bo
  //  16..50 g1 {Wc,Wgc1,Wgc2,Wgc3,bc, gat1[10], gat2[10], gat3[10]}
  //  51..85 g2 (same layout)
  const float* feat1 = (const float*)d_in[0];
  const float* adj1 = (const float*)d_in[1];
  const float* bias1 = (const float*)d_in[2];
  const float* feat2 = (const float*)d_in[3];
  const float* adj2 = (const float*)d_in[4];
  const float* bias2 = (const float*)d_in[5];
  const int* GID1 = (const int*)d_in[6];
  const int* GID2 = (const int*)d_in[7];
  const float* Wd1 = (const float*)d_in[8];
  const float* Wf = (const float*)d_in[9];
  const float* Wlm = (const float*)d_in[10];
  const float* Wo = (const float*)d_in[11];
  const float* bd1 = (const float*)d_in[12];
  const float* bfv = (const float*)d_in[13];
  const float* blm = (const float*)d_in[14];
  const float* bo = (const float*)d_in[15];
  GParams g1 = load_g(d_in, 16);
  GParams g2 = load_g(d_in, 51);

  float* out = (float*)d_out;
  Ctx c{stream, (char*)d_ws, 0};

  // shared spgat scratch, sized for the worst case (Fin padded = 2048, N=2048)
  const int FM = 2048, NN = 2048;
  SpBufs S;
  S.WT = (bf16*)c.take((size_t)8 * 64 * FM * 2);
  S.WresT = (bf16*)c.take((size_t)8 * 64 * FM * 2);
  S.fts = (float*)c.take((size_t)8 * NN * 64 * 4);
  S.ftsT = (bf16*)c.take((size_t)8 * 64 * NN * 2);
  S.resid = (float*)c.take((size_t)8 * NN * 64 * 4);
  S.f1 = (float*)c.take((size_t)8 * NN * 4);
  S.f2 = (float*)c.take((size_t)8 * NN * 4);
  S.h1 = (float*)c.take((size_t)NN * 512 * 4);
  S.h1bf = (bf16*)c.take((size_t)NN * 512 * 2);
  S.BWT = (bf16*)c.take((size_t)8 * 64 * 512 * 2);
  S.BWresT = (bf16*)c.take((size_t)8 * 64 * 512 * 2);
  S.fts2 = (float*)c.take((size_t)8 * NN * 64 * 4);
  S.fts2T = (bf16*)c.take((size_t)8 * 64 * NN * 2);
  S.resid2 = (float*)c.take((size_t)8 * NN * 64 * 4);
  S.f1b = (float*)c.take((size_t)8 * NN * 4);
  S.f2b = (float*)c.take((size_t)8 * NN * 4);
  S.hB = (float*)c.take((size_t)8 * NN * 64 * 4);

  float *emb1, *cancat1, *emb2, *cancat2;
  bf16 *emb1bf, *emb2bf;
  branch(c, S, feat1, adj1, bias1, g1, N1, F1, out, &emb1, &emb1bf, &cancat1);
  branch(c, S, feat2, adj2, bias2, g2, N2, F2, out + (long)N1 * N1,
         &emb2, &emb2bf, &cancat2);

  // ---- head ----
  bf16* Wd1T = (bf16*)c.take((size_t)64 * 128 * 2);
  bf16* WlmT = (bf16*)c.take((size_t)64 * 64 * 2);
  bf16* WfT = (bf16*)c.take((size_t)128 * 128 * 2);
  bf16* WoT = (bf16*)c.take((size_t)64 * 128 * 2);
  float* d1 = (float*)c.take((size_t)N1 * 64 * 4);
  bf16* d1bf = (bf16*)c.take((size_t)N1 * 64 * 2);
  float* latent = (float*)c.take((size_t)N1 * 64 * 4);
  bf16* mebf = (bf16*)c.take((size_t)M * 128 * 2);
  float* fbuf = (float*)c.take((size_t)M * 128 * 4);
  bf16* fbf = (bf16*)c.take((size_t)M * 128 * 2);

  conv(c, Wd1, 0, Wd1T, 0, 64, 64, 128, 128, 1, 0, 1);
  gemm(c, emb1bf, 128, 0, Wd1T, 128, 0, d1, 64, 0, bd1,
       d1bf, 64, 0, nullptr, 0, 0, N1, 64, 128, 1, 1);
  conv(c, Wlm, 0, WlmT, 0, 64, 64, 64, 64, 1, 0, 1);
  gemm(c, d1bf, 64, 0, WlmT, 64, 0, latent, 64, 0, blm,
       nullptr, 0, 0, nullptr, 0, 0, N1, 64, 64, 0, 1);
  k_gather_me<<<dim3((unsigned)((M * 128 + 255) / 256)), dim3(256), 0, stream>>>(
      latent, cancat2, GID1, GID2, mebf, M);
  conv(c, Wf, 0, WfT, 0, 128, 128, 128, 128, 1, 0, 1);
  gemm(c, mebf, 128, 0, WfT, 128, 0, fbuf, 128, 0, bfv,
       fbf, 128, 0, nullptr, 0, 0, M, 128, 128, 1, 1);
  // Wo is [128][2]; WoT padded with zero rows to 64 so the GEMM runs branch-free
  conv(c, Wo, 0, WoT, 0, 2, 64, 128, 128, 1, 0, 1);
  gemm(c, fbf, 128, 0, WoT, 128, 0, out + (long)N1 * N1 + (long)N2 * N2, 2, 0, bo,
       nullptr, 0, 0, nullptr, 0, 0, M, 2, 128, 0, 1);
  (void)cancat1; (void)emb2; (void)emb2bf;
}
